// VectorQuantizerEMA_19232863552081
// MI455X (gfx1250) — matrix operand order repaired and checked by reading, not compiled, not run on hardware
//
#include <hip/hip_runtime.h>
#include <hip/hip_bf16.h>
#include <math.h>

typedef __attribute__((ext_vector_type(16))) __bf16 v16bf;
typedef __attribute__((ext_vector_type(8)))  float  v8f;
typedef __attribute__((ext_vector_type(4)))  int    v4i;

// ---------------------------------------------------------------------------
// Async global->LDS copy of one 16-byte segment (CDNA5 GLOBAL_LOAD_ASYNC_TO_LDS,
// ASYNCcnt).  Builtin signature (from hipcc diagnostics):
//   (v4i addrspace(1)* src, v4i addrspace(3)* dst, imm offset, imm cpol)
// Guarded: falls back to a register-relay copy if the builtin is unavailable.
// ---------------------------------------------------------------------------
#if __has_builtin(__builtin_amdgcn_global_load_async_to_lds_b128) && \
    __has_builtin(__builtin_amdgcn_s_wait_asynccnt)
#define VQ_HAVE_ASYNC_LDS 1
#else
#define VQ_HAVE_ASYNC_LDS 0
#endif

__device__ __forceinline__ void vq_copy16(const void* g, void* l) {
#if VQ_HAVE_ASYNC_LDS
    __builtin_amdgcn_global_load_async_to_lds_b128(
        (__attribute__((address_space(1))) v4i*)g,
        (__attribute__((address_space(3))) v4i*)l, 0, 0);
#else
    *(uint4*)l = *(const uint4*)g;        // global_load_b128 + ds_store_b128
#endif
}

__device__ __forceinline__ void vq_wait_async() {
#if VQ_HAVE_ASYNC_LDS
    __builtin_amdgcn_s_wait_asynccnt(0);
#endif
}

// ---------------------------------------------------------------------------
// Kernel 1: per-code squared norms + scatter emb_w into bf16 hi/lo arrays laid
// out exactly as the WMMA B operand (32x16 bf16 tile per (code-tile, k-chunk)):
//   lane = col + 16*(k%32 / 16), element e = k%16, so each lane owns 16
//   contiguous bf16 (32B).
// ---------------------------------------------------------------------------
__global__ void vq_prep(const float* __restrict__ emb, float* __restrict__ e_norm,
                        __bf16* __restrict__ bhi, __bf16* __restrict__ blo, int D) {
    const int code = blockIdx.x;      // one block per code, blockDim.x == D == 256
    const int d    = threadIdx.x;
    const float v  = emb[(size_t)code * D + d];

    __shared__ float red[256];
    red[d] = v * v;
    __syncthreads();
    for (int s = 128; s > 0; s >>= 1) {
        if (d < s) red[d] += red[d + s];
        __syncthreads();
    }
    if (d == 0) e_norm[code] = red[0];

    // split fp32 -> bf16 hi + bf16 lo (hi rounds RNE, lo captures residual)
    __bf16 hi = (__bf16)v;
    __bf16 lo = (__bf16)(v - (float)hi);

    const int t    = code >> 4;       // code tile (16 codes)
    const int col  = code & 15;       // B column within tile
    const int c    = d >> 5;          // k-chunk of 32
    const int kr   = d & 31;
    const int half = kr >> 4;
    const int e    = kr & 15;
    const int lane = col + 16 * half;
    const size_t o = ((size_t)(t * 8 + c) * 32 + lane) * 16 + e;
    bhi[o] = hi;
    blo[o] = lo;
}

// ---------------------------------------------------------------------------
// Kernel 2: WMMA argmin.  8 waves / block; each wave owns M=32 rows (two A
// register tiles), so each 16KB B tile staged in LDS feeds 48 WMMAs per wave.
// B tiles are double-buffered in LDS via async global->LDS copies overlapped
// with the WMMA chain of the current tile (one workgroup barrier per tile).
// e_norm is staged once into LDS so the tile loop has no global dependency.
// A operand = (-2*z) as bf16 hi/lo in contiguous k order (k = 16*half + e); the
// accumulator is initialized to ||e_n||^2 so acc = dist - ||z_m||^2 directly.
// ---------------------------------------------------------------------------
__global__ void __launch_bounds__(256)
vq_argmin(const float* __restrict__ z, const __bf16* __restrict__ bhi,
          const __bf16* __restrict__ blo, const float* __restrict__ e_norm,
          int* __restrict__ idx_out, float* __restrict__ counts, int K) {
    // [buffer][hi/lo][8 chunks * 32 lanes * 16 bf16] = 2*2*8KB = 32KB LDS
    __shared__ __attribute__((aligned(64))) __bf16 lds_b[2][2][4096];
    __shared__ __attribute__((aligned(64))) float  lds_en[1024];   // K <= 1024

    const int tid  = threadIdx.x;
    const int lane = tid & 31;
    const int wave = tid >> 5;
    const int m    = lane & 15;
    const int half = lane >> 4;
    const int rowBase = blockIdx.x * 256 + wave * 32;

    // ---- load A (32 rows x 256 k) into registers as -2*z split bf16 ----
    v16bf a_hi[2][8], a_lo[2][8];
    #pragma unroll
    for (int s = 0; s < 2; ++s) {
        const float* zr = z + (size_t)(rowBase + s * 16 + m) * 256;
        #pragma unroll
        for (int c = 0; c < 8; ++c) {
            const int k0 = 32 * c + 16 * half;
            #pragma unroll
            for (int e = 0; e < 8; ++e) {
                float x1 = -2.0f * zr[k0 + e];           // k = 32c + 16*half + e
                __bf16 h1 = (__bf16)x1;
                a_hi[s][c][e]     = h1;
                a_lo[s][c][e]     = (__bf16)(x1 - (float)h1);
                float x2 = -2.0f * zr[k0 + 8 + e];       // k = 32c + 16*half + 8 + e
                __bf16 h2 = (__bf16)x2;
                a_hi[s][c][e + 8] = h2;
                a_lo[s][c][e + 8] = (__bf16)(x2 - (float)h2);
            }
        }
    }

    float bestv[2][8];
    int   besti[2][8];
    #pragma unroll
    for (int s = 0; s < 2; ++s)
        #pragma unroll
        for (int r = 0; r < 8; ++r) { bestv[s][r] = 3.0e38f; besti[s][r] = 0; }

    const int tiles = K >> 4;                             // 64 code tiles

    // stage tile 0 into buffer 0 + the whole e_norm vector (4KB) into LDS
    {
        #pragma unroll
        for (int j = 0; j < 2; ++j) {
            const int seg = tid * 2 + j;                  // 0..511 (x 8 bf16)
            vq_copy16(bhi + seg * 8, &lds_b[0][0][seg * 8]);
            vq_copy16(blo + seg * 8, &lds_b[0][1][seg * 8]);
        }
        if (tid * 4 < K) vq_copy16(e_norm + tid * 4, &lds_en[tid * 4]);
    }

    for (int t = 0; t < tiles; ++t) {
        vq_wait_async();
        __syncthreads();          // tile t visible; buf[(t+1)&1] free to reuse

        if (t + 1 < tiles) {      // overlap next tile's copy with this compute
            const __bf16* gh = bhi + (size_t)(t + 1) * 4096;
            const __bf16* gl = blo + (size_t)(t + 1) * 4096;
            __bf16* dh = &lds_b[(t + 1) & 1][0][0];
            __bf16* dl = &lds_b[(t + 1) & 1][1][0];
            #pragma unroll
            for (int j = 0; j < 2; ++j) {
                const int seg = tid * 2 + j;
                vq_copy16(gh + seg * 8, dh + seg * 8);
                vq_copy16(gl + seg * 8, dl + seg * 8);
            }
        }

        const float en = lds_en[t * 16 + m];              // ds_load_b32, short latency
        v8f acc0, acc1;
        #pragma unroll
        for (int r = 0; r < 8; ++r) { acc0[r] = en; acc1[r] = en; }

        const v16bf* bh = (const v16bf*)&lds_b[t & 1][0][0];
        const v16bf* bl = (const v16bf*)&lds_b[t & 1][1][0];
        #pragma unroll
        for (int c = 0; c < 8; ++c) {
            v16bf Bh = bh[c * 32 + lane];                 // 2x ds_load_b128
            v16bf Bl = bl[c * 32 + lane];
            acc0 = __builtin_amdgcn_wmma_f32_16x16x32_bf16(false, a_hi[0][c], false, Bh,
                                                           (short)0, acc0, false, false);
            acc0 = __builtin_amdgcn_wmma_f32_16x16x32_bf16(false, a_hi[0][c], false, Bl,
                                                           (short)0, acc0, false, false);
            acc0 = __builtin_amdgcn_wmma_f32_16x16x32_bf16(false, a_lo[0][c], false, Bh,
                                                           (short)0, acc0, false, false);
            acc1 = __builtin_amdgcn_wmma_f32_16x16x32_bf16(false, a_hi[1][c], false, Bh,
                                                           (short)0, acc1, false, false);
            acc1 = __builtin_amdgcn_wmma_f32_16x16x32_bf16(false, a_hi[1][c], false, Bl,
                                                           (short)0, acc1, false, false);
            acc1 = __builtin_amdgcn_wmma_f32_16x16x32_bf16(false, a_lo[1][c], false, Bh,
                                                           (short)0, acc1, false, false);
        }

        const int code = t * 16 + m;
        #pragma unroll
        for (int r = 0; r < 8; ++r) {
            float s0 = acc0[r];
            if (s0 < bestv[0][r]) { bestv[0][r] = s0; besti[0][r] = code; }
            float s1 = acc1[r];
            if (s1 < bestv[1][r]) { bestv[1][r] = s1; besti[1][r] = code; }
        }
    }

    // ---- reduce across the 16 columns of each half-wave (C layout:
    //      lanes 0-15 hold M=r, lanes 16-31 hold M=8+r) ----
    #pragma unroll
    for (int s = 0; s < 2; ++s) {
        #pragma unroll
        for (int r = 0; r < 8; ++r) {
            float v = bestv[s][r];
            int   i = besti[s][r];
            #pragma unroll
            for (int mask = 8; mask >= 1; mask >>= 1) {
                float ov = __shfl_xor(v, mask, 16);
                int   oi = __shfl_xor(i, mask, 16);
                if (ov < v || (ov == v && oi < i)) { v = ov; i = oi; }
            }
            bestv[s][r] = v;
            besti[s][r] = i;
        }
    }

    if (m == 0) {                                         // lanes 0 and 16 write
        #pragma unroll
        for (int s = 0; s < 2; ++s) {
            const int row = rowBase + s * 16 + half * 8;
            #pragma unroll
            for (int r = 0; r < 8; ++r) {
                idx_out[row + r] = besti[s][r];
                atomicAdd(&counts[besti[s][r]], 1.0f);    // exact integer-valued adds
            }
        }
    }
}

// ---------------------------------------------------------------------------
// Kernel 3: gather quantized rows, write straight-through output, and emit a
// deterministic per-block partial sum of (q - z)^2.
// ---------------------------------------------------------------------------
__global__ void vq_gather(const float* __restrict__ z, const float* __restrict__ emb,
                          const int* __restrict__ idx, float* __restrict__ outq,
                          float* __restrict__ partials) {
    const int b   = blockIdx.x;       // 128 rows per block
    const int tid = threadIdx.x;      // 256 threads
    float acc = 0.0f;
    for (int i = tid; i < 128 * 256; i += 256) {
        const int n = b * 128 + (i >> 8);
        const int d = i & 255;
        const int code = idx[n];
        const float q  = emb[(size_t)code * 256 + d];
        const float dv = q - z[(size_t)n * 256 + d];
        outq[(size_t)n * 256 + d] = q;                // straight-through == quantized
        acc += dv * dv;
    }
    __shared__ float red[256];
    red[tid] = acc;
    __syncthreads();
    for (int s = 128; s > 0; s >>= 1) {
        if (tid < s) red[tid] += red[tid + s];
        __syncthreads();
    }
    if (tid == 0) partials[b] = red[0];
}

// ---------------------------------------------------------------------------
// Kernel 4: final scalars.  entropy over K counts + sum of nb partials.
// ---------------------------------------------------------------------------
__global__ void vq_finalize(const float* __restrict__ counts,
                            const float* __restrict__ partials,
                            float* __restrict__ loss_out, float* __restrict__ ppl_out,
                            int K, int nb, float invN, float invND) {
    const int t = threadIdx.x;        // 1024 threads
    __shared__ float redA[1024];
    __shared__ float redB[1024];
    float ent = 0.0f, ss = 0.0f;
    if (t < K) {
        const float p = counts[t] * invN;
        ent = p * logf(p + 1e-10f);
    }
    if (t < nb) ss = partials[t];
    redA[t] = ent;
    redB[t] = ss;
    __syncthreads();
    for (int s = 512; s > 0; s >>= 1) {
        if (t < s) { redA[t] += redA[t + s]; redB[t] += redB[t + s]; }
        __syncthreads();
    }
    if (t == 0) {
        *ppl_out  = expf(-redA[0]);
        *loss_out = 0.25f * (redB[0] * invND);
    }
}

// ---------------------------------------------------------------------------
extern "C" void kernel_launch(void* const* d_in, const int* in_sizes, int n_in,
                              void* d_out, int out_size, void* d_ws, size_t ws_size,
                              hipStream_t stream) {
    const float* z_e = (const float*)d_in[0];
    const float* emb = (const float*)d_in[1];
    float* out = (float*)d_out;

    const int D = 256;
    const int N = in_sizes[0] / D;    // 131072
    const int K = in_sizes[1] / D;    // 1024
    const int nb = N / 128;           // 1024 row-blocks (gather)

    // workspace carve-up (256B aligned)
    char* ws = (char*)d_ws;
    size_t off = 0;
    float* e_norm   = (float*)(ws + off); off += ((size_t)K * 4 + 255) & ~(size_t)255;
    float* counts   = (float*)(ws + off); off += ((size_t)K * 4 + 255) & ~(size_t)255;
    float* partials = (float*)(ws + off); off += ((size_t)nb * 4 + 255) & ~(size_t)255;
    __bf16* bhi     = (__bf16*)(ws + off); off += ((size_t)K * D * 2 + 255) & ~(size_t)255;
    __bf16* blo     = (__bf16*)(ws + off); off += ((size_t)K * D * 2 + 255) & ~(size_t)255;
    int* idx        = (int*)(ws + off);   off += ((size_t)N * 4 + 255) & ~(size_t)255;

    // output layout: [loss][quantized_st N*D][perplexity]
    float* loss_out = out;
    float* outq     = out + 1;
    float* ppl_out  = out + 1 + (size_t)N * D;

    (void)hipMemsetAsync(counts, 0, (size_t)K * sizeof(float), stream);

    vq_prep<<<K, D, 0, stream>>>(emb, e_norm, bhi, blo, D);
    vq_argmin<<<N / 256, 256, 0, stream>>>(z_e, bhi, blo, e_norm, idx, counts, K);
    vq_gather<<<nb, 256, 0, stream>>>(z_e, emb, idx, outq, partials);
    vq_finalize<<<1, 1024, 0, stream>>>(counts, partials, loss_out, ppl_out,
                                        K, nb, 1.0f / (float)N,
                                        1.0f / ((float)N * (float)D));
}
